// GraphAttentionLayer_35356170781255
// MI455X (gfx1250) — matrix operand order repaired and checked by reading, not compiled, not run on hardware
//
#include <hip/hip_runtime.h>
#include <hip/hip_bf16.h>

// ---------------------------------------------------------------------------
// GraphAttentionLayer for MI455X (gfx1250, wave32, WMMA)
//   h:[4,4096,128] f32, adj:[4096,4096] i32, W:[128,64] f32, a:[128,1] f32
//   out:[4,4096,64] f32
// v3: fix async-to-LDS builtin pointer types (v4i addrspace(1)* / (3)*),
//     discovered from the clang diagnostic in round 2.
// ---------------------------------------------------------------------------

typedef __bf16 bf16_t;
typedef __attribute__((ext_vector_type(16))) __bf16 bf16x16;
typedef __attribute__((ext_vector_type(8)))  __bf16 bf16x8;
typedef __attribute__((ext_vector_type(8)))  float  f32x8;
typedef int gat_v4i __attribute__((__vector_size__(16)));

#define ALPHA   0.2f
#define NEG_BIG (-9.0e15f)
#define B_SZ    4
#define N_SZ    4096
#define FIN     128
#define FOUT    64

#if __has_builtin(__builtin_amdgcn_global_load_async_to_lds_b128)
#define GAT_ASYNC_LDS 1
#define GAT_GPTR(p) ((__attribute__((address_space(1))) gat_v4i*)(p))
#define GAT_LPTR(p) ((__attribute__((address_space(3))) gat_v4i*)(p))
#else
#define GAT_ASYNC_LDS 0
#endif

__device__ __forceinline__ float lrelu(float x) { return x > 0.0f ? x : ALPHA * x; }

// 16-bit A-matrix 16x32 fragment (ISA 7.12.2): lane holds row M = lane&15.
// element e -> K offset = abase + (e<8 ? e : e+8), abase = (lane&16)?8:0.
// The 16 elements are two contiguous runs: [abase, abase+8) and
// [abase+16, abase+24)  -> 4x b128 vector loads.
// 16-bit B-matrix 32x16 fragment: lane holds column N = lane&15;
// K offset same as A: abase + (e<8 ? e : e+8) -> two 16B loads.

// ---------------------------------------------------------------------------
// K0: Wt[o][k] = bf16(W[k][o])   (one-shot 64x128 transpose)
// ---------------------------------------------------------------------------
__global__ void gat_wt_kernel(const float* __restrict__ W, bf16_t* __restrict__ Wt) {
    const int i = blockIdx.x * blockDim.x + threadIdx.x;
    if (i >= FOUT * FIN) return;
    const int o = i / FIN, k = i % FIN;
    Wt[i] = (bf16_t)W[k * FOUT + o];
}

// ---------------------------------------------------------------------------
// K1: hW = h @ W  (one wave -> 16 rows x 64 cols), bf16 WMMA, K=128.
// Stores hw_f32 row-major (for f1/f2) and hwT bf16 transposed [b][o][n].
// ---------------------------------------------------------------------------
__global__ void gat_hw_kernel(const float* __restrict__ h,
                              const bf16_t* __restrict__ Wt,
                              float* __restrict__ hw_f32,
                              bf16_t* __restrict__ hwT) {
    const int wave = blockIdx.x * (blockDim.x >> 5) + (threadIdx.x >> 5);
    const int lane = threadIdx.x & 31;
    const int tilesPerBatch = N_SZ / 16;
    const int b  = wave / tilesPerBatch;
    const int n0 = (wave % tilesPerBatch) * 16;

    const float* hrow = h + ((size_t)b * N_SZ + n0 + (lane & 15)) * FIN;
    const int abase = (lane & 16) ? 8 : 0;
    const int bbase = (lane & 16) ? 16 : 0;

    f32x8 c[4] = {};
    for (int k0 = 0; k0 < FIN; k0 += 32) {
        alignas(16) float hv[16];
        *(float4*)&hv[0]  = *(const float4*)(hrow + k0 + abase);
        *(float4*)&hv[4]  = *(const float4*)(hrow + k0 + abase + 4);
        *(float4*)&hv[8]  = *(const float4*)(hrow + k0 + abase + 16);
        *(float4*)&hv[12] = *(const float4*)(hrow + k0 + abase + 20);
        bf16x16 afrag;
#pragma unroll
        for (int e = 0; e < 16; ++e) afrag[e] = (bf16_t)hv[e];
#pragma unroll
        for (int ot = 0; ot < 4; ++ot) {
            const int col = ot * 16 + (lane & 15);
            const bf16x8 blo = *(const bf16x8*)(Wt + col * FIN + k0 + abase);
            const bf16x8 bhi = *(const bf16x8*)(Wt + col * FIN + k0 + abase + 16);
            const bf16x16 bfrag = __builtin_shufflevector(blo, bhi, 0, 1, 2, 3, 4, 5, 6, 7, 8, 9, 10, 11, 12, 13, 14, 15);
            c[ot] = __builtin_amdgcn_wmma_f32_16x16x32_bf16(
                false, afrag, false, bfrag, (short)0, c[ot], false, false);
        }
    }
    // C/D layout: lane holds N = lane&15; VGPR r holds M = r + (lane>=16 ? 8:0)
    const int off = (lane & 16) ? 8 : 0;
#pragma unroll
    for (int ot = 0; ot < 4; ++ot) {
        const int o = ot * 16 + (lane & 15);
        bf16x8 v;
#pragma unroll
        for (int r = 0; r < 8; ++r) {
            const int n = n0 + r + off;
            hw_f32[((size_t)b * N_SZ + n) * FOUT + o] = c[ot][r];
            v[r] = (bf16_t)c[ot][r];
        }
        *(bf16x8*)(hwT + ((size_t)b * FOUT + o) * N_SZ + n0 + off) = v;  // 16B store
    }
}

// ---------------------------------------------------------------------------
// K2: f1[r] = hW[r,:] . a[:64],  f2[r] = hW[r,:] . a[64:]
// ---------------------------------------------------------------------------
__global__ void gat_f_kernel(const float* __restrict__ hw,
                             const float* __restrict__ a,
                             float* __restrict__ f1, float* __restrict__ f2) {
    const int idx = blockIdx.x * blockDim.x + threadIdx.x;
    if (idx >= B_SZ * N_SZ) return;
    const float* row = hw + (size_t)idx * FOUT;
    float s1 = 0.0f, s2 = 0.0f;
#pragma unroll
    for (int o = 0; o < FOUT; o += 4) {
        const float4 v  = *(const float4*)(row + o);
        const float4 a1 = *(const float4*)(a + o);
        const float4 a2 = *(const float4*)(a + FOUT + o);
        s1 += v.x * a1.x + v.y * a1.y + v.z * a1.z + v.w * a1.w;
        s2 += v.x * a2.x + v.y * a2.y + v.z * a2.z + v.w * a2.w;
    }
    f1[idx] = s1;
    f2[idx] = s2;
}

// ---------------------------------------------------------------------------
// K3: per-batch max of f2 (lrelu monotone -> rowmax = lrelu(f1 + maxf2))
// ---------------------------------------------------------------------------
__global__ void gat_maxf2_kernel(const float* __restrict__ f2,
                                 float* __restrict__ maxf2) {
    __shared__ float red[256];
    const int b = blockIdx.x;
    float m = -3.0e38f;
    for (int i = threadIdx.x; i < N_SZ; i += blockDim.x)
        m = fmaxf(m, f2[b * N_SZ + i]);
    red[threadIdx.x] = m;
    __syncthreads();
    for (int s = 128; s > 0; s >>= 1) {
        if ((int)threadIdx.x < s)
            red[threadIdx.x] = fmaxf(red[threadIdx.x], red[threadIdx.x + s]);
        __syncthreads();
    }
    if (threadIdx.x == 0) maxf2[b] = red[0];
}

// ---------------------------------------------------------------------------
// K4: per-row softmax denominator (one wave32 per row)
// ---------------------------------------------------------------------------
__global__ void gat_rowsum_kernel(const float* __restrict__ f1,
                                  const float* __restrict__ f2,
                                  const float* __restrict__ maxf2,
                                  float* __restrict__ rowmax,
                                  float* __restrict__ invsum) {
    const int wave = blockIdx.x * (blockDim.x >> 5) + (threadIdx.x >> 5);
    const int lane = threadIdx.x & 31;
    if (wave >= B_SZ * N_SZ) return;
    const int b = wave / N_SZ;
    const float f1v = f1[wave];
    const float rm  = lrelu(f1v + maxf2[b]);
    const float* f2b = f2 + (size_t)b * N_SZ;
    float s = 0.0f;
    for (int m = lane * 4; m < N_SZ; m += 128) {
        const float4 v = *(const float4*)(f2b + m);
        s += __expf(lrelu(f1v + v.x) - rm);
        s += __expf(lrelu(f1v + v.y) - rm);
        s += __expf(lrelu(f1v + v.z) - rm);
        s += __expf(lrelu(f1v + v.w) - rm);
    }
#pragma unroll
    for (int off = 16; off > 0; off >>= 1)
        s += __shfl_xor(s, off, 32);
    if (lane == 0) {
        rowmax[wave] = rm;
        invsum[wave] = 1.0f / s;
    }
}

// ---------------------------------------------------------------------------
// K5: out = sigmoid( where(adj>0, softmax(e), NEG_BIG) @ hW )
// 8 waves/block share one 32x64 bf16 B-tile staged in LDS per K chunk
// (GLOBAL_LOAD_ASYNC_TO_LDS_B128, ASYNCcnt). 32 blocks/batch -> b uniform.
// ---------------------------------------------------------------------------
__global__ void __launch_bounds__(256)
gat_attn_kernel(const bf16_t* __restrict__ hwT,
                const int* __restrict__ adj,
                const float* __restrict__ f1,
                const float* __restrict__ f2,
                const float* __restrict__ rowmax,
                const float* __restrict__ invsum,
                float* __restrict__ out) {
    __shared__ alignas(16) bf16_t tile[FOUT * 32];   // [o][m] 64 x 32, 4 KB

    const int tid  = threadIdx.x;
    const int lane = tid & 31;
    const int wave = blockIdx.x * (blockDim.x >> 5) + (tid >> 5);
    const int tilesPerBatch = N_SZ / 16;             // 256
    const int b  = wave / tilesPerBatch;
    const int n0 = (wave % tilesPerBatch) * 16;

    // Staging assignment: 64 o-rows x 4 chunks of 8 bf16 (16B) = 256 threads
    const int so = tid >> 2;
    const int sp = (tid & 3) * 8;
    const bf16_t* gsrc = hwT + ((size_t)b * FOUT + so) * N_SZ + sp;
    bf16_t* ldst = &tile[so * 32 + sp];

    const int arow = n0 + (lane & 15);
    const size_t ridx = (size_t)b * N_SZ + arow;
    const float f1v = f1[ridx];
    const float rm  = rowmax[ridx];
    const float is  = invsum[ridx];
    const int*   adjrow = adj + (size_t)arow * N_SZ;
    const float* f2b    = f2 + (size_t)b * N_SZ;

    const int abase = (lane & 16) ? 8 : 0;
    const int bbase = (lane & 16) ? 16 : 0;

    f32x8 c[4] = {};
    for (int m0 = 0; m0 < N_SZ; m0 += 32) {
        // ---- stage B tile: hwT[b][0:64][m0:m0+32] -> LDS -------------------
#if GAT_ASYNC_LDS
        __builtin_amdgcn_global_load_async_to_lds_b128(
            GAT_GPTR(gsrc + m0), GAT_LPTR(ldst), 0, 0);
#if __has_builtin(__builtin_amdgcn_s_wait_asynccnt)
        __builtin_amdgcn_s_wait_asynccnt(0);
#else
        asm volatile("s_wait_asynccnt 0" ::: "memory");
#endif
#else
        *(uint4*)ldst = *(const uint4*)(gsrc + m0);
#endif
        __syncthreads();

        // ---- A fragment: attention values generated in-register ------------
        alignas(16) float f2v[16];
        alignas(16) int   adv[16];
        *(float4*)&f2v[0]  = *(const float4*)(f2b + m0 + abase);
        *(float4*)&f2v[4]  = *(const float4*)(f2b + m0 + abase + 4);
        *(float4*)&f2v[8]  = *(const float4*)(f2b + m0 + abase + 16);
        *(float4*)&f2v[12] = *(const float4*)(f2b + m0 + abase + 20);
        *(int4*)&adv[0]    = *(const int4*)(adjrow + m0 + abase);
        *(int4*)&adv[4]    = *(const int4*)(adjrow + m0 + abase + 4);
        *(int4*)&adv[8]    = *(const int4*)(adjrow + m0 + abase + 16);
        *(int4*)&adv[12]   = *(const int4*)(adjrow + m0 + abase + 20);
        bf16x16 afrag;
#pragma unroll
        for (int e = 0; e < 16; ++e) {
            const float ev  = lrelu(f1v + f2v[e]);
            const float att = (adv[e] > 0) ? (__expf(ev - rm) * is) : NEG_BIG;
            afrag[e] = (bf16_t)att;
        }

        // ---- 4 WMMAs against the shared LDS tile ---------------------------
#pragma unroll
        for (int ot = 0; ot < 4; ++ot) {
            const int col = ot * 16 + (lane & 15);
            const bf16x8 blo = *(const bf16x8*)&tile[col * 32 + abase];
            const bf16x8 bhi = *(const bf16x8*)&tile[col * 32 + abase + 16];
            const bf16x16 bfrag = __builtin_shufflevector(blo, bhi, 0, 1, 2, 3, 4, 5, 6, 7, 8, 9, 10, 11, 12, 13, 14, 15);
            c[ot] = __builtin_amdgcn_wmma_f32_16x16x32_bf16(
                false, afrag, false, bfrag, (short)0, c[ot], false, false);
        }
        __syncthreads();
    }

    const int off = (lane & 16) ? 8 : 0;
#pragma unroll
    for (int ot = 0; ot < 4; ++ot) {
        const int o = ot * 16 + (lane & 15);
#pragma unroll
        for (int r = 0; r < 8; ++r) {
            const int n = n0 + r + off;
            const float v = c[ot][r];
            out[((size_t)b * N_SZ + n) * FOUT + o] = 1.0f / (1.0f + __expf(-v));
        }
    }
}

// ---------------------------------------------------------------------------
extern "C" void kernel_launch(void* const* d_in, const int* in_sizes, int n_in,
                              void* d_out, int out_size, void* d_ws, size_t ws_size,
                              hipStream_t stream) {
    (void)in_sizes; (void)n_in; (void)out_size; (void)ws_size;
    const float* h   = (const float*)d_in[0];
    const int*   adj = (const int*)d_in[1];
    const float* W   = (const float*)d_in[2];
    const float* a   = (const float*)d_in[3];
    float* out = (float*)d_out;

    // Workspace carve-up (all offsets 256B-aligned)
    char* ws = (char*)d_ws;
    float*  hw_f32 = (float*)(ws);                 // 4 MB  [B,N,64] f32
    bf16_t* hwT    = (bf16_t*)(ws + 4194304);      // 2 MB  [B,64,N] bf16 (transposed)
    bf16_t* Wt     = (bf16_t*)(ws + 6291456);      // 16 KB [64,128] bf16 (transposed)
    float*  f1     = (float*)(ws + 6307840);       // 64 KB
    float*  f2     = (float*)(ws + 6373376);       // 64 KB
    float*  rowmax = (float*)(ws + 6438912);       // 64 KB
    float*  invsum = (float*)(ws + 6504448);       // 64 KB
    float*  maxf2  = (float*)(ws + 6569984);       // 16 B

    gat_wt_kernel<<<(FOUT * FIN) / 256, 256, 0, stream>>>(W, Wt);

    const int waves_hw = B_SZ * N_SZ / 16;         // 1024 tiles
    gat_hw_kernel<<<waves_hw / 8, 256, 0, stream>>>(h, Wt, hw_f32, hwT);

    gat_f_kernel<<<(B_SZ * N_SZ) / 256, 256, 0, stream>>>(hw_f32, a, f1, f2);

    gat_maxf2_kernel<<<B_SZ, 256, 0, stream>>>(f2, maxf2);

    gat_rowsum_kernel<<<(B_SZ * N_SZ) / 8, 256, 0, stream>>>(f1, f2, maxf2,
                                                             rowmax, invsum);

    gat_attn_kernel<<<waves_hw / 8, 256, 0, stream>>>(hwT, adj, f1, f2,
                                                      rowmax, invsum, out);
}
